// PAIR_MODULE_16372415332367
// MI455X (gfx1250) — matrix operand order repaired and checked by reading, not compiled, not run on hardware
//
#include <hip/hip_runtime.h>
#include <hip/hip_bf16.h>

typedef __attribute__((ext_vector_type(16))) _Float16 v16h;
typedef __attribute__((ext_vector_type(8)))  _Float16 v8h;
typedef __attribute__((ext_vector_type(8)))  float    v8f;

union H16 { v16h v; v8h h[2]; };

#define NBATCH 1024
#define NR 10
#define NH 512
#define N2H 1024
#define NROWS 10240                 // NBATCH*NR
#define WMAT_ELEMS (NH * N2H)       // 524288 halves per transposed weight matrix

// workspace layout (float indices unless noted)
#define V_OFF 0                     // v = W1@W2, 1024 floats (v_h = [0:512], v_q = [512:1024])
#define C_OFF 1024                  // scalar c = b1@W2 + b2
#define SH_OFF 1040                 // sh[10240]
#define SQ_OFF 11280                // sq[10240]
#define WT_BYTE_OFF 131072          // f16 transposed weights: 4 * 524288 halves = 4 MB

// ---------------------------------------------------------------------------
// Kernel 1: v = W1 @ W2, c = b1@W2 + b2, and zero sh/sq accumulators.
// ---------------------------------------------------------------------------
__global__ void prep_kernel(const float* __restrict__ W1, const float* __restrict__ b1,
                            const float* __restrict__ W2, const float* __restrict__ b2,
                            float* __restrict__ wsf) {
  int idx = blockIdx.x * blockDim.x + threadIdx.x;
  if (idx < N2H) {
    float s = 0.f;
    for (int m = 0; m < NH; ++m) s += W1[idx * NH + m] * W2[m];
    wsf[V_OFF + idx] = s;
  } else if (idx == N2H) {
    float s = b2[0];
    for (int m = 0; m < NH; ++m) s += b1[m] * W2[m];
    wsf[C_OFF] = s;
  } else {
    int z = idx - (N2H + 1);
    if (z < 2 * NROWS) wsf[SH_OFF + z] = 0.f;   // zero sh and sq (contiguous)
  }
}

// ---------------------------------------------------------------------------
// Kernel 2: convert + transpose the four 1024x512 f32 weights into f16
// [N=512][K=1024] row-major so WMMA B-fragments are contiguous loads.
// ---------------------------------------------------------------------------
__global__ void wt_kernel(const float* __restrict__ Why, const float* __restrict__ Whg,
                          const float* __restrict__ Wqy, const float* __restrict__ Wqg,
                          _Float16* __restrict__ wt) {
  int idx = blockIdx.x * blockDim.x + threadIdx.x;
  if (idx >= 4 * WMAT_ELEMS) return;
  int mat = idx >> 19;
  int rem = idx & (WMAT_ELEMS - 1);
  int n = rem >> 10;       // 0..511 (output column of original W)
  int k = rem & 1023;      // 0..1023
  const float* W = (mat == 0) ? Why : (mat == 1) ? Whg : (mat == 2) ? Wqy : Wqg;
  wt[idx] = (_Float16)W[k * NH + n];
}

// ---------------------------------------------------------------------------
// Kernel 3: fused gated-transform GEMMs (WMMA f16 -> f32) + dot with v.
// Block = 16 rows of hist or ques, 8 waves each owning 64 columns for BOTH
// the tanh (Wy) and leaky-relu (Wg) branches.
// ---------------------------------------------------------------------------
__global__ void __launch_bounds__(256) gated_dot_kernel(
    const float* __restrict__ hist, const float* __restrict__ ques,
    const _Float16* __restrict__ wt,
    const float* __restrict__ bhy, const float* __restrict__ bhg,
    const float* __restrict__ bqy, const float* __restrict__ bqg,
    const float* __restrict__ vbase,   // wsf + V_OFF
    float* __restrict__ sh, float* __restrict__ sq) {
  // 16 rows x 1024 K of f16; row stride padded +8 halves (16B) -> conflict-free b128 LDS loads
  __shared__ _Float16 smem[16 * 1032];

  const bool isHist = blockIdx.x < (NROWS / 16);
  const int  row0   = (blockIdx.x % (NROWS / 16)) * 16;
  const float*    X   = isHist ? hist : ques;
  const _Float16* Wy  = wt + (size_t)(isHist ? 0 : 2) * WMAT_ELEMS;
  const _Float16* Wg  = Wy + WMAT_ELEMS;
  const float*    by  = isHist ? bhy : bqy;
  const float*    bg  = isHist ? bhg : bqg;
  const float*    vv  = vbase + (isHist ? 0 : NH);
  float*          sOut = isHist ? sh : sq;

  const int tid = threadIdx.x;
  // cooperative f32 -> f16 staging of the 16x1024 row tile (coalesced)
  for (int i = tid; i < 16 * N2H; i += 256) {
    int r = i >> 10, k = i & 1023;
    smem[r * 1032 + k] = (_Float16)X[(size_t)(row0 + r) * N2H + k];
  }
  __syncthreads();

  const int wv = tid >> 5, lane = tid & 31;
  const int lr = lane & 15, hif = lane >> 4;
  const int nb = wv * 64;                       // this wave's column base (of 512)
  const _Float16* lrow = smem + lr * 1032;      // A row for this lane (M = lane&15)

  v8f ay[4] = {};
  v8f ag[4] = {};

  for (int kt = 0; kt < 32; ++kt) {
    const int k0 = kt * 32;
    // A fragment (16x32 f16): lanes 0-15 K=0..15, lanes 16-31 K=16..31 (contiguous)
    H16 a;
    a.h[0] = *(const v8h*)(lrow + k0 + hif * 16);
    a.h[1] = *(const v8h*)(lrow + k0 + hif * 16 + 8);
#pragma unroll
    for (int nt = 0; nt < 4; ++nt) {
      const int col = nb + nt * 16 + lr;        // B column = lane&15
      // B fragment (32x16 f16): lanes 0-15 K=0..15, lanes 16-31 K=16..31 (contiguous 32B)
      const _Float16* byp = Wy + (size_t)col * N2H + k0 + hif * 16;
      H16 b;
      b.h[0] = *(const v8h*)(byp);
      b.h[1] = *(const v8h*)(byp + 8);
      ay[nt] = __builtin_amdgcn_wmma_f32_16x16x32_f16(false, a.v, false, b.v,
                                                      (short)0, ay[nt], false, false);
      const _Float16* bgp = Wg + (size_t)col * N2H + k0 + hif * 16;
      b.h[0] = *(const v8h*)(bgp);
      b.h[1] = *(const v8h*)(bgp + 8);
      ag[nt] = __builtin_amdgcn_wmma_f32_16x16x32_f16(false, a.v, false, b.v,
                                                      (short)0, ag[nt], false, false);
    }
  }

  // gate: tanh(y + by) * leaky_relu(g + bg, 0.01), then dot with v per row
  float p[8];
#pragma unroll
  for (int r = 0; r < 8; ++r) p[r] = 0.f;
#pragma unroll
  for (int nt = 0; nt < 4; ++nt) {
    const int col = nb + nt * 16 + lr;          // C layout: N = lane&15
    const float byv = by[col], bgv = bg[col], vs = vv[col];
#pragma unroll
    for (int r = 0; r < 8; ++r) {               // C layout: M = r + (lane>=16 ? 8 : 0)
      float yv = ay[nt][r] + byv;
      float gv = ag[nt][r] + bgv;
      float val = tanhf(yv) * (gv >= 0.f ? gv : 0.01f * gv);
      p[r] += val * vs;
    }
  }
  // reduce across the 16 lanes of each half-wave (same row, different columns)
#pragma unroll
  for (int m = 1; m < 16; m <<= 1)
#pragma unroll
    for (int r = 0; r < 8; ++r) p[r] += __shfl_xor(p[r], m, 32);

  if (lr == 0) {
    const int rb = row0 + hif * 8;
#pragma unroll
    for (int r = 0; r < 8; ++r) atomicAdd(&sOut[rb + r], p[r]);
  }
}

// ---------------------------------------------------------------------------
// Kernel 4: per-(b,i) argmax of masked logits+gumbel -> one-hot output.
// ---------------------------------------------------------------------------
__global__ void score_kernel(const float* __restrict__ sh, const float* __restrict__ sq,
                             const float* __restrict__ cptr, const float* __restrict__ gumbel,
                             const float* __restrict__ Watt, const float* __restrict__ batt,
                             float* __restrict__ out) {
  int idx = blockIdx.x * blockDim.x + threadIdx.x;
  if (idx >= NBATCH * NR) return;
  int b = idx / NR, i = idx % NR;
  float c = cptr[0], w0 = Watt[0], w1 = Watt[1], bb = batt[0];
  float sqi = sq[idx];
  float best = -3.0e38f;
  int bj = 0;
  for (int j = 0; j < NR; ++j) {
    if (j <= i) {
      float logit = (sh[b * NR + j] + sqi + c) * w0 + (float)(i - j + 1) * w1 + bb
                    + gumbel[idx * NR + j];
      if (logit > best) { best = logit; bj = j; }   // strict > keeps first max
    }
  }
  for (int j = 0; j < NR; ++j) out[idx * NR + j] = (j == bj) ? 1.0f : 0.0f;
}

// ---------------------------------------------------------------------------
extern "C" void kernel_launch(void* const* d_in, const int* in_sizes, int n_in,
                              void* d_out, int out_size, void* d_ws, size_t ws_size,
                              hipStream_t stream) {
  (void)in_sizes; (void)n_in; (void)out_size; (void)ws_size;
  const float* hist   = (const float*)d_in[0];
  const float* ques   = (const float*)d_in[1];
  const float* gumbel = (const float*)d_in[2];
  const float* Why    = (const float*)d_in[3];
  const float* bhy    = (const float*)d_in[4];
  const float* Whg    = (const float*)d_in[5];
  const float* bhg    = (const float*)d_in[6];
  const float* Wqy    = (const float*)d_in[7];
  const float* bqy    = (const float*)d_in[8];
  const float* Wqg    = (const float*)d_in[9];
  const float* bqg    = (const float*)d_in[10];
  const float* W1     = (const float*)d_in[11];
  const float* b1     = (const float*)d_in[12];
  const float* W2     = (const float*)d_in[13];
  const float* b2     = (const float*)d_in[14];
  const float* Watt   = (const float*)d_in[15];
  const float* batt   = (const float*)d_in[16];

  float*     wsf = (float*)d_ws;
  _Float16*  wt  = (_Float16*)((char*)d_ws + WT_BYTE_OFF);
  float*     out = (float*)d_out;

  // 1) v, c, zero accumulators (re-done every call: graph-replay safe)
  prep_kernel<<<85, 256, 0, stream>>>(W1, b1, W2, b2, wsf);
  // 2) f16 transposed weights
  wt_kernel<<<8192, 256, 0, stream>>>(Why, Whg, Wqy, Wqg, wt);
  // 3) fused WMMA gated transforms + dot with v  (1280 blocks: 640 hist + 640 ques)
  gated_dot_kernel<<<1280, 256, 0, stream>>>(hist, ques, wt, bhy, bhg, bqy, bqg,
                                             wsf + V_OFF, wsf + SH_OFF, wsf + SQ_OFF);
  // 4) argmax one-hot
  score_kernel<<<40, 256, 0, stream>>>(wsf + SH_OFF, wsf + SQ_OFF, wsf + C_OFF,
                                       gumbel, Watt, batt, out);
}
